// HierarchicalGraphNet_45251775431203
// MI455X (gfx1250) — matrix operand order repaired and checked by reading, not compiled, not run on hardware
//
#include <hip/hip_runtime.h>
#include <hip/hip_bf16.h>
#include <math.h>

// ---------------------------------------------------------------------------
// HierarchicalGraphNet on MI455X (gfx1250, wave32, WMMA)
// N=100000 nodes, D=256, L=3 layers, E=320000 edges, G=4096 graphs
// GEMMs on v_wmma_f32_16x16x32_f16; GRU GEMMs fused with gates+LayerNorm,
// keeping gi/gh in LDS (96KB of the 320KB WGP LDS) to avoid 1.2GB/layer of
// HBM traffic. BatchNorm scale/shift folded into the gi GEMM A-conversion.
// ---------------------------------------------------------------------------

typedef __attribute__((ext_vector_type(16))) _Float16 v16h;
typedef __attribute__((ext_vector_type(8)))  float    v8f;

#define NN_  100000
#define EE_  320000
#define DD_  256
#define LL_  3
#define GG_  4096
#define REP_ 1280   // (L+2)*D

// ---------------- small utility kernels ----------------

__global__ __launch_bounds__(256) void k_copy_f32(const float* __restrict__ src,
                                                  float* __restrict__ dst, int n) {
  int i = blockIdx.x * 256 + threadIdx.x;
  if (i < n) dst[i] = src[i];
}

__global__ __launch_bounds__(256) void k_cvt_f16(const float* __restrict__ src,
                                                 _Float16* __restrict__ dst, int n) {
  int i = blockIdx.x * 256 + threadIdx.x;
  if (i < n) dst[i] = (_Float16)src[i];
}

__global__ __launch_bounds__(256) void k_fill(float* __restrict__ p, float v, int n) {
  int i = blockIdx.x * 256 + threadIdx.x;
  if (i < n) p[i] = v;
}

// ---------------- GIN scatter-add: agg[dst] += h[src] ----------------
__global__ __launch_bounds__(256) void k_scatter_add(const float* __restrict__ h,
                                                     const int* __restrict__ ei,
                                                     float* __restrict__ agg) {
  int idx = blockIdx.x * 256 + threadIdx.x;        // E*256 threads exactly
  int e = idx >> 8;
  int c = idx & 255;
  int s = ei[e];
  int d = ei[EE_ + e];
  atomicAdd(agg + (size_t)d * DD_ + c, h[(size_t)s * DD_ + c]);
}

// ---------------- WMMA GEMM: C = act(A[M,Din] * W[Dout,Din]^T + bias) -------
// block = 8 waves; wave w owns rows r0 = bx*128 + w*16, col group c0 = by*64.
__global__ __launch_bounds__(256)
void k_gemm_wmma(const float* __restrict__ A, const _Float16* __restrict__ W,
                 const float* __restrict__ bias, float* __restrict__ C,
                 int M, int Din, int Dout, int relu) {
  const int wave = threadIdx.x >> 5;
  const int lane = threadIdx.x & 31;
  const int half = lane >> 4;
  const int l16  = lane & 15;
  const int r0 = blockIdx.x * 128 + wave * 16;
  const int c0 = blockIdx.y * 64;
  if (r0 >= M) return;                       // uniform per wave -> EXEC stays full

  int rowA = r0 + l16;
  if (rowA >= M) rowA = M - 1;               // clamp edge-tile reads
  const float* aBase = A + (size_t)rowA * Din;

  const _Float16* w0 = W + (size_t)(c0 +  0 + l16) * Din;
  const _Float16* w1 = W + (size_t)(c0 + 16 + l16) * Din;
  const _Float16* w2 = W + (size_t)(c0 + 32 + l16) * Din;
  const _Float16* w3 = W + (size_t)(c0 + 48 + l16) * Din;

  v8f acc0 = {}, acc1 = {}, acc2 = {}, acc3 = {};

  for (int k0 = 0; k0 < Din; k0 += 32) {
    __builtin_prefetch(aBase + k0 + 64, 0, 3);
    v8f alo = *(const v8f*)(aBase + k0 + half * 16);
    v8f ahi = *(const v8f*)(aBase + k0 + half * 16 + 8);
    v16h a;
#pragma unroll
    for (int i = 0; i < 8; ++i) {
      a[i]     = (_Float16)alo[i];
      a[8 + i] = (_Float16)ahi[i];
    }
    const int kb = k0 + half * 16;
    v16h b0 = *(const v16h*)(w0 + kb);
    v16h b1 = *(const v16h*)(w1 + kb);
    v16h b2 = *(const v16h*)(w2 + kb);
    v16h b3 = *(const v16h*)(w3 + kb);

    acc0 = __builtin_amdgcn_wmma_f32_16x16x32_f16(false, a, false, b0, (short)0, acc0, false, false);
    acc1 = __builtin_amdgcn_wmma_f32_16x16x32_f16(false, a, false, b1, (short)0, acc1, false, false);
    acc2 = __builtin_amdgcn_wmma_f32_16x16x32_f16(false, a, false, b2, (short)0, acc2, false, false);
    acc3 = __builtin_amdgcn_wmma_f32_16x16x32_f16(false, a, false, b3, (short)0, acc3, false, false);
  }

  v8f accs[4] = {acc0, acc1, acc2, acc3};
#pragma unroll
  for (int t = 0; t < 4; ++t) {
    const int col = c0 + t * 16 + l16;
    const float bv = bias[col];
#pragma unroll
    for (int r = 0; r < 8; ++r) {
      int row = r0 + r + half * 8;
      if (row < M) {
        float v = accs[t][r] + bv;
        if (relu) v = v > 0.0f ? v : 0.0f;
        C[(size_t)row * Dout + col] = v;
      }
    }
  }
}

// ---------------- batch-norm stats over N (per feature) ----------------

__global__ __launch_bounds__(256) void k_bn_stats(const float* __restrict__ g,
                                                  float* __restrict__ sum,
                                                  float* __restrict__ sumsq,
                                                  int N, int rowsPerBlock) {
  const int c = threadIdx.x;
  int r0 = blockIdx.x * rowsPerBlock;
  int r1 = r0 + rowsPerBlock; if (r1 > N) r1 = N;
  float s = 0.0f, s2 = 0.0f;
  for (int r = r0; r < r1; ++r) {
    float v = g[(size_t)r * DD_ + c];
    s += v; s2 += v * v;
  }
  atomicAdd(&sum[c], s);
  atomicAdd(&sumsq[c], s2);
}

__global__ __launch_bounds__(256) void k_bn_finalize(const float* __restrict__ sum,
                                                     const float* __restrict__ sumsq,
                                                     const float* __restrict__ bng,
                                                     const float* __restrict__ bnb,
                                                     float* __restrict__ scale,
                                                     float* __restrict__ shift, int N) {
  int c = threadIdx.x;
  float inv = 1.0f / (float)N;
  float mu = sum[c] * inv;
  float var = sumsq[c] * inv - mu * mu;
  float sc = bng[c] * rsqrtf(var + 1e-5f);
  scale[c] = sc;
  shift[c] = bnb[c] - mu * sc;
}

// ---------------- fused GRU: (BN(g)@Wih^T) & (h@Whh^T) in LDS, then --------
// gates + LayerNorm + running m/e. Block = 8 waves = one 16-node tile;
// wave w computes cols [w*96, w*96+96) of both 768-wide GEMMs (6 WMMA accs
// each), dumps to LDS (2*16*768 f32 = 96KB), then 256 threads apply gates.
#define GRU_SMEM_FLOATS (2 * 16 * 768 + 256)

__global__ __launch_bounds__(256)
void k_gru_fused(const float* __restrict__ G, const float* __restrict__ H,
                 const _Float16* __restrict__ Wih, const _Float16* __restrict__ Whh,
                 const float* __restrict__ bih, const float* __restrict__ bhh,
                 const float* __restrict__ bnscale, const float* __restrict__ bnshift,
                 const float* __restrict__ lng, const float* __restrict__ lnb,
                 float* __restrict__ hout, float* __restrict__ mbuf,
                 float* __restrict__ ebuf, int first) {
  extern __shared__ float smem[];
  float* sg   = smem;                  // gi: [16][768]
  float* sh   = smem + 16 * 768;      // gh: [16][768]
  float* sred = smem + 2 * 16 * 768;  // [256] LN reduction

  const int wave = threadIdx.x >> 5;
  const int lane = threadIdx.x & 31;
  const int half = lane >> 4;
  const int l16  = lane & 15;
  const int r0   = blockIdx.x * 16;   // N % 16 == 0: no edge handling needed
  const int c0w  = wave * 96;

  // ---- phase 1a: gi = (g*bnscale + bnshift) @ Wih^T + bih ----
  {
    const float* aBase = G + (size_t)(r0 + l16) * 256;
    v8f acc[6] = {};
    for (int k0 = 0; k0 < 256; k0 += 32) {
      v8f alo = *(const v8f*)(aBase + k0 + half * 16);
      v8f ahi = *(const v8f*)(aBase + k0 + half * 16 + 8);
      v8f slo = *(const v8f*)(bnscale + k0 + half * 16);
      v8f shi = *(const v8f*)(bnscale + k0 + half * 16 + 8);
      v8f tlo = *(const v8f*)(bnshift + k0 + half * 16);
      v8f thi = *(const v8f*)(bnshift + k0 + half * 16 + 8);
      v16h a;
#pragma unroll
      for (int i = 0; i < 8; ++i) {
        a[i]     = (_Float16)(alo[i] * slo[i] + tlo[i]);
        a[8 + i] = (_Float16)(ahi[i] * shi[i] + thi[i]);
      }
      const int kb = k0 + half * 16;
#pragma unroll
      for (int t = 0; t < 6; ++t) {
        v16h b = *(const v16h*)(Wih + (size_t)(c0w + t * 16 + l16) * 256 + kb);
        acc[t] = __builtin_amdgcn_wmma_f32_16x16x32_f16(false, a, false, b, (short)0, acc[t], false, false);
      }
    }
#pragma unroll
    for (int t = 0; t < 6; ++t) {
      int col = c0w + t * 16 + l16;
      float bv = bih[col];
#pragma unroll
      for (int r = 0; r < 8; ++r)
        sg[(r + 8 * half) * 768 + col] = acc[t][r] + bv;
    }
  }

  // ---- phase 1b: gh = h @ Whh^T + bhh ----
  {
    const float* aBase = H + (size_t)(r0 + l16) * 256;
    v8f acc[6] = {};
    for (int k0 = 0; k0 < 256; k0 += 32) {
      v8f alo = *(const v8f*)(aBase + k0 + half * 16);
      v8f ahi = *(const v8f*)(aBase + k0 + half * 16 + 8);
      v16h a;
#pragma unroll
      for (int i = 0; i < 8; ++i) {
        a[i]     = (_Float16)alo[i];
        a[8 + i] = (_Float16)ahi[i];
      }
      const int kb = k0 + half * 16;
#pragma unroll
      for (int t = 0; t < 6; ++t) {
        v16h b = *(const v16h*)(Whh + (size_t)(c0w + t * 16 + l16) * 256 + kb);
        acc[t] = __builtin_amdgcn_wmma_f32_16x16x32_f16(false, a, false, b, (short)0, acc[t], false, false);
      }
    }
#pragma unroll
    for (int t = 0; t < 6; ++t) {
      int col = c0w + t * 16 + l16;
      float bv = bhh[col];
#pragma unroll
      for (int r = 0; r < 8; ++r)
        sh[(r + 8 * half) * 768 + col] = acc[t][r] + bv;
    }
  }
  __syncthreads();

  // ---- phase 2: gates + LayerNorm per node ----
  const int c = threadIdx.x;
  const float lgc = lng[c], lbc = lnb[c];
  for (int n = 0; n < 16; ++n) {
    const size_t b1 = (size_t)(r0 + n) * 256 + c;
    float ir  = sg[n * 768 + c],       hr = sh[n * 768 + c];
    float iz  = sg[n * 768 + 256 + c], hz = sh[n * 768 + 256 + c];
    float in_ = sg[n * 768 + 512 + c], hn = sh[n * 768 + 512 + c];
    float r = 1.0f / (1.0f + __expf(-(ir + hr)));
    float z = 1.0f / (1.0f + __expf(-(iz + hz)));
    float nn = tanhf(in_ + r * hn);
    float hv = (1.0f - z) * nn + z * H[b1];

    sred[c] = hv; __syncthreads();
#pragma unroll
    for (int s = 128; s > 0; s >>= 1) {
      if (c < s) sred[c] += sred[c + s];
      __syncthreads();
    }
    float mu = sred[0] * (1.0f / 256.0f);
    __syncthreads();
    float d = hv - mu;
    sred[c] = d * d; __syncthreads();
#pragma unroll
    for (int s = 128; s > 0; s >>= 1) {
      if (c < s) sred[c] += sred[c + s];
      __syncthreads();
    }
    float var = sred[0] * (1.0f / 256.0f);
    __syncthreads();   // all reads of sred done before next iteration writes

    float o = d * rsqrtf(var + 1e-5f) * lgc + lbc;
    hout[b1] = o;
    if (first) { mbuf[b1] = o; ebuf[b1] = o; }
    else       { mbuf[b1] *= o; ebuf[b1] += o; }
  }
}

// ---------------- segment max (atomic float-max trick) ----------------

__device__ __forceinline__ void atomicMaxF(float* addr, float v) {
  if (v >= 0.0f) atomicMax((int*)addr, __float_as_int(v));
  else           atomicMin((unsigned int*)addr, __float_as_uint(v));
}

__global__ __launch_bounds__(256) void k_seg_max(const float* __restrict__ v,
                                                 const int* __restrict__ batch,
                                                 float* __restrict__ out, int slice) {
  int n = blockIdx.x, c = threadIdx.x;
  int g = batch[n];
  atomicMaxF(out + (size_t)g * REP_ + slice * DD_ + c, v[(size_t)n * DD_ + c]);
}

// ---------------------------------------------------------------------------

extern "C" void kernel_launch(void* const* d_in, const int* in_sizes, int n_in,
                              void* d_out, int out_size, void* d_ws, size_t ws_size,
                              hipStream_t stream) {
  const float* x       = (const float*)d_in[0];
  const float* lin1_w  = (const float*)d_in[1];
  const float* lin1_b  = (const float*)d_in[2];
  const float* lin2_w  = (const float*)d_in[3];
  const float* lin2_b  = (const float*)d_in[4];
  const float* bn_g    = (const float*)d_in[5];
  const float* bn_b    = (const float*)d_in[6];
  const float* gru_wih = (const float*)d_in[7];
  const float* gru_whh = (const float*)d_in[8];
  const float* gru_bih = (const float*)d_in[9];
  const float* gru_bhh = (const float*)d_in[10];
  const float* ln_g    = (const float*)d_in[11];
  const float* ln_b    = (const float*)d_in[12];
  const int*   edge_ix = (const int*)d_in[13];
  const int*   batch   = (const int*)d_in[14];
  float* out = (float*)d_out;

  const int N = NN_, D = DD_, G = GG_;
  const size_t ND = (size_t)N * D;

  // workspace carve-out (256B aligned)
  char* ws = (char*)d_ws;
  size_t off = 0;
  auto carve = [&](size_t bytes) -> void* {
    void* p = ws + off;
    off = (off + bytes + 255) & ~(size_t)255;
    return p;
  };
  _Float16* wf_lin1 = (_Float16*)carve((size_t)LL_ * D * D * sizeof(_Float16));
  _Float16* wf_lin2 = (_Float16*)carve((size_t)LL_ * D * D * sizeof(_Float16));
  _Float16* wf_wih  = (_Float16*)carve((size_t)LL_ * 3 * D * D * sizeof(_Float16));
  _Float16* wf_whh  = (_Float16*)carve((size_t)LL_ * 3 * D * D * sizeof(_Float16));
  float* hA   = (float*)carve(ND * 4);   // h ping
  float* hB   = (float*)carve(ND * 4);   // h pong
  float* agg  = (float*)carve(ND * 4);   // agg, then reused as g
  float* tbuf = (float*)carve(ND * 4);
  float* mbuf = (float*)carve(ND * 4);
  float* ebuf = (float*)carve(ND * 4);
  float* bnsum   = (float*)carve(D * 4);
  float* bnsumsq = (float*)carve(D * 4);
  float* bnscale = (float*)carve(D * 4);
  float* bnshift = (float*)carve(D * 4);

  // ---- per-call prep: weight f32->f16, out = -inf ----
  {
    int c1 = LL_ * D * D;       // 196608
    int c3 = LL_ * 3 * D * D;   // 589824
    k_cvt_f16<<<(c1 + 255) / 256, 256, 0, stream>>>(lin1_w, wf_lin1, c1);
    k_cvt_f16<<<(c1 + 255) / 256, 256, 0, stream>>>(lin2_w, wf_lin2, c1);
    k_cvt_f16<<<(c3 + 255) / 256, 256, 0, stream>>>(gru_wih, wf_wih, c3);
    k_cvt_f16<<<(c3 + 255) / 256, 256, 0, stream>>>(gru_whh, wf_whh, c3);
    int no = G * REP_;          // 5,242,880
    k_fill<<<(no + 255) / 256, 256, 0, stream>>>(out, -INFINITY, no);
  }

  const dim3 blk(256);
  const dim3 gemm_grid_small((N + 127) / 128, D / 64);
  const int scatterBlocks = (EE_ * D) / 256;
  const int bnRows = 256;
  const int bnBlocks = (N + bnRows - 1) / bnRows;
  const int gruBlocks = N / 16;                        // 6250 (exact)
  const size_t gruSmem = GRU_SMEM_FLOATS * sizeof(float);  // 99,328 B (<320KB WGP LDS)

  const float* hin = x;
  for (int i = 0; i < LL_; ++i) {
    float* hcur = (i & 1) ? hB : hA;   // ping-pong

    // 1) GIN aggregation: agg = h + segment_sum(h[src] -> dst)
    k_copy_f32<<<(int)((ND + 255) / 256), 256, 0, stream>>>(hin, agg, (int)ND);
    k_scatter_add<<<scatterBlocks, blk, 0, stream>>>(hin, edge_ix, agg);

    // 2) t = relu(agg @ lin1_w[i].T + lin1_b[i])
    k_gemm_wmma<<<gemm_grid_small, blk, 0, stream>>>(
        agg, wf_lin1 + (size_t)i * D * D, lin1_b + (size_t)i * D, tbuf, N, D, D, 1);

    // 3) g = relu(t @ lin2_w[i].T + lin2_b[i])  (into agg buffer)
    k_gemm_wmma<<<gemm_grid_small, blk, 0, stream>>>(
        tbuf, wf_lin2 + (size_t)i * D * D, lin2_b + (size_t)i * D, agg, N, D, D, 1);

    // 4) batch-norm stats -> scale/shift (apply folded into fused GRU GEMM)
    k_fill<<<1, 256, 0, stream>>>(bnsum, 0.0f, D);
    k_fill<<<1, 256, 0, stream>>>(bnsumsq, 0.0f, D);
    k_bn_stats<<<bnBlocks, blk, 0, stream>>>(agg, bnsum, bnsumsq, N, bnRows);
    k_bn_finalize<<<1, 256, 0, stream>>>(bnsum, bnsumsq, bn_g + (size_t)i * D,
                                         bn_b + (size_t)i * D, bnscale, bnshift, N);

    // 5) fused: gi/gh GEMMs (LDS-resident) + GRU gates + LayerNorm + m/e
    k_gru_fused<<<gruBlocks, blk, gruSmem, stream>>>(
        agg, hin, wf_wih + (size_t)i * 3 * D * D, wf_whh + (size_t)i * 3 * D * D,
        gru_bih + (size_t)i * 3 * D, gru_bhh + (size_t)i * 3 * D,
        bnscale, bnshift, ln_g, ln_b, hcur, mbuf, ebuf, i == 0 ? 1 : 0);

    // 6) incremental segment-max of h_i into out slice i
    k_seg_max<<<N, blk, 0, stream>>>(hcur, batch, out, i);

    hin = hcur;
  }

  // m and e slices
  k_seg_max<<<N, blk, 0, stream>>>(mbuf, batch, out, LL_);
  k_seg_max<<<N, blk, 0, stream>>>(ebuf, batch, out, LL_ + 1);
}
